// CrossAttention_FactorizedAVInteractions_20538533609566
// MI455X (gfx1250) — matrix operand order repaired and checked by reading, not compiled, not run on hardware
//
#include <hip/hip_runtime.h>
#include <hip/hip_bf16.h>

// ---------------------------------------------------------------------------
// Types for CDNA5 WMMA (wave32): v16bf A/B fragments, v8f accumulator.
// ---------------------------------------------------------------------------
typedef __attribute__((ext_vector_type(16))) __bf16 bf16x16;
typedef __attribute__((ext_vector_type(8)))  __bf16 bf16x8;
typedef __attribute__((ext_vector_type(8)))  float  f32x8;
typedef __attribute__((ext_vector_type(4)))  float  f32x4;

union AFragU { bf16x16 v; bf16x8 h[2]; };

// --- CDNA5 data-mover feature detection -----------------------------------
#if __has_builtin(__builtin_amdgcn_tensor_load_to_lds) && \
    __has_builtin(__builtin_amdgcn_s_wait_tensorcnt)
#define USE_TDM 1
#else
#define USE_TDM 0
#endif

#if __has_builtin(__builtin_amdgcn_global_load_async_to_lds_b128)
#define USE_ASYNC_LDS 1
#else
#define USE_ASYNC_LDS 0
#endif

typedef unsigned int tdm_v4u __attribute__((ext_vector_type(4)));
typedef int          tdm_v8i __attribute__((ext_vector_type(8)));
typedef int          tdm_v4i __attribute__((ext_vector_type(4)));

typedef __attribute__((address_space(3))) void lds_void_t;
typedef __attribute__((address_space(1))) void gbl_void_t;

__device__ __forceinline__ unsigned lds_offset_of(void* p) {
    // addrspace(3) pointers are the raw 32-bit LDS byte offset
    return (unsigned)(size_t)(lds_void_t*)p;
}

#define BM 128
#define BN 128
#define BK 32

// ---------------------------------------------------------------------------
// Generic tiled WMMA GEMM:  C(M,N) = A(M,K) * BT(N,K)^T  [+ bias]
//   A  : fp32, row-remapped (phys_row = (r/agrp)*astride + aoff + r%agrp),
//        converted fp32->bf16 while staging into LDS.
//   BT : bf16, PRE-TRANSPOSED weights (N x K row-major) -> the LDS tile
//        Bs[n][k] is a contiguous 2-D tile copy, moved by the Tensor Data
//        Mover (TENSOR_LOAD_TO_LDS, one 16-row tile per wave) when available,
//        else per-lane async LDS copies, else plain b128 copies.
//   Output: optional fp32 (Cf) and/or bf16 (Cb).
// Requires M%128==0, N%128==0, K%32==0 (true for every GEMM here).
// Block = 256 threads = 8 waves; wave (2x4 grid) owns a 64x32 C tile (4x2 WMMA).
// ---------------------------------------------------------------------------
__global__ __launch_bounds__(256) void gemm_wmma_bf16(
    const float* __restrict__ A, const __bf16* __restrict__ BT,
    int M, int N, int K,
    int agrp, int astride, int aoff,
    const float* __restrict__ bias,
    float* __restrict__ Cf, __bf16* __restrict__ Cb)
{
    __shared__ alignas(64) __bf16 As[BM][BK];   // row-major 128x32 bf16 (8KB)
    __shared__ alignas(64) __bf16 Bs[BN][BK];   // Bs[n][k] (8KB)

    const int tid  = threadIdx.x;
    const int wave = tid >> 5;
    const int lane = tid & 31;
    const int bm = blockIdx.x * BM;
    const int bn = blockIdx.y * BN;
    const int wm = (wave >> 2) * 64;   // wave row offset in C tile
    const int wn = (wave & 3) * 32;    // wave col offset in C tile

    f32x8 acc[4][2];
#pragma unroll
    for (int i = 0; i < 4; ++i)
#pragma unroll
        for (int j = 0; j < 2; ++j)
#pragma unroll
            for (int e = 0; e < 8; ++e) acc[i][j][e] = 0.0f;

    // A loader: 256 threads cover 128x32 fp32 -> 16 contiguous elems/thread
    const int ar = tid >> 1;            // 0..127
    const int ac = (tid & 1) << 4;      // 0 or 16
    const int arow = bm + ar;
    const int prow = (arow / agrp) * astride + aoff + (arow % agrp);
    const float* abase = A + (size_t)prow * K;

#if USE_TDM
    // Per-wave TDM descriptor: copy 16 rows x 32 bf16 of BT into Bs[wave*16..]
    const unsigned lds_base = lds_offset_of(&Bs[wave * 16][0]);
#endif

    for (int k0 = 0; k0 < K; k0 += BK) {
        __syncthreads();   // previous iteration's LDS reads complete

        // ---- stage A tile (fp32 -> bf16) ----
        {
            const float* s = abase + k0 + ac;
#pragma unroll
            for (int t = 0; t < 16; ++t) As[ar][ac + t] = (__bf16)s[t];
        }

        // ---- stage B tile: contiguous rows of pre-transposed bf16 weights ----
#if USE_TDM
        {
            // D# per CDNA5 ISA 8.3/8.4: 2-D tile, data_size=2B (code 1),
            // tile_dim0=32 (K-contig), tile_dim1=16 rows, dim0_stride=K.
            const unsigned long long ga =
                (unsigned long long)(const void*)(BT + (size_t)(bn + wave * 16) * K + k0);
            tdm_v4u g0;
            g0[0] = 1u;                                   // count=1, user mode
            g0[1] = lds_base;                             // lds_addr
            g0[2] = (unsigned)ga;                         // global_addr[31:0]
            g0[3] = ((unsigned)(ga >> 32) & 0x1FFFFFFu) | (2u << 30); // [56:32]|type=2
            tdm_v8i g1;
            g1[0] = 0x10000;                              // data_size=1 (2B), mask=0
            g1[1] = (int)(((unsigned)K & 0xFFFFu) << 16); // tensor_dim0 lo16
            g1[2] = (int)(((unsigned)K >> 16) | (((unsigned)N & 0xFFFFu) << 16));
            g1[3] = (int)(((unsigned)N >> 16) | (32u << 16));   // tile_dim0=32
            g1[4] = 16;                                   // tile_dim1=16, tile_dim2=0
            g1[5] = (int)K;                               // tensor_dim0_stride lo32
            g1[6] = 0;                                    // stride hi | dim1_stride lo
            g1[7] = 0;
            tdm_v4i gz4 = {0, 0, 0, 0};
            tdm_v8i gz8 = {0, 0, 0, 0, 0, 0, 0, 0};
            __builtin_amdgcn_tensor_load_to_lds(g0, g1, gz4, gz4, gz8, 0);
            __builtin_amdgcn_s_wait_tensorcnt(0);
        }
#elif USE_ASYNC_LDS
        {
            // per-lane async 16B copies: 512 chunks, 2 per thread
#pragma unroll
            for (int it = 0; it < 2; ++it) {
                const int c  = tid + 256 * it;
                const int n  = c >> 2;
                const int ko = (c & 3) << 3;
                const __bf16* g = BT + (size_t)(bn + n) * K + k0 + ko;
                __builtin_amdgcn_global_load_async_to_lds_b128(
                    (gbl_void_t*)(void*)g, (lds_void_t*)(void*)&Bs[n][ko], 0, 0);
            }
#if __has_builtin(__builtin_amdgcn_s_wait_asynccnt)
            __builtin_amdgcn_s_wait_asynccnt(0);
#else
            asm volatile("s_wait_asynccnt 0" ::: "memory");
#endif
        }
#else
        {
#pragma unroll
            for (int it = 0; it < 2; ++it) {
                const int c  = tid + 256 * it;
                const int n  = c >> 2;
                const int ko = (c & 3) << 3;
                *(bf16x8*)&Bs[n][ko] =
                    *(const bf16x8*)(BT + (size_t)(bn + n) * K + k0 + ko);
            }
        }
#endif
        __syncthreads();

        // prefetch next A tile rows (global_prefetch_b8)
        if (k0 + BK < K) {
            __builtin_prefetch(abase + k0 + BK + ac, 0, 1);
        }

        // ---- fragments in contiguous k order (k = 16*(lane/16) + s), same for A and B ----
        // A 16x32: lane l holds row (l&15); lanes<16 K in {0..15},
        // lanes>=16 K in {16..31}  -> two contiguous 8-elem chunks.
        const int r  = lane & 15;
        const int kh = (lane >> 4) << 4;     // 0 or 16
        AFragU af[4];
#pragma unroll
        for (int i = 0; i < 4; ++i) {
            const __bf16* p = &As[wm + 16 * i + r][0];
            af[i].h[0] = *(const bf16x8*)(p + kh);
            af[i].h[1] = *(const bf16x8*)(p + kh + 8);
        }
        // B 32x16: lane l holds column (l&15); lanes<16 K=0..15, lanes>=16
        // K=16..31 -> 16 contiguous bf16 in Bs[n][k].
        const int kb = (lane >> 4) << 4;     // 0 or 16
        bf16x16 bf[2];
#pragma unroll
        for (int j = 0; j < 2; ++j)
            bf[j] = *(const bf16x16*)&Bs[wn + 16 * j + (lane & 15)][kb];

        // ---- 8 WMMAs per wave per K-step ----
#pragma unroll
        for (int i = 0; i < 4; ++i)
#pragma unroll
            for (int j = 0; j < 2; ++j)
                acc[i][j] = __builtin_amdgcn_wmma_f32_16x16x32_bf16(
                    false, af[i].v, false, bf[j], (short)0, acc[i][j],
                    false, false);
    }

    // ---- epilogue: C 16x16 layout -> lanes<16: M=vgpr, lanes>=16: M=8+vgpr
    const int cn0 = lane & 15;
    const int mh  = (lane >> 4) << 3;
#pragma unroll
    for (int i = 0; i < 4; ++i) {
#pragma unroll
        for (int j = 0; j < 2; ++j) {
            const int n = bn + wn + 16 * j + cn0;
            const float bv = bias ? bias[n] : 0.0f;
#pragma unroll
            for (int v = 0; v < 8; ++v) {
                const int m = bm + wm + 16 * i + mh + v;
                const float val = acc[i][j][v] + bv;
                if (Cf) Cf[(size_t)m * N + n] = val;
                if (Cb) Cb[(size_t)m * N + n] = (__bf16)val;
            }
        }
    }
}

// ---------------------------------------------------------------------------
// Weight pre-pass: W (K x N fp32, row-major) -> WT (N x K bf16, row-major)
// ---------------------------------------------------------------------------
__global__ void transpose_to_bf16_kernel(const float* __restrict__ W,
                                         __bf16* __restrict__ WT,
                                         int K, int N, int total)
{
    for (int g = blockIdx.x * blockDim.x + threadIdx.x; g < total;
         g += gridDim.x * blockDim.x) {
        const int n = g / K, k = g % K;
        WT[g] = (__bf16)W[(size_t)k * N + n];
    }
}

// ---------------------------------------------------------------------------
// Cross-attention core (N1=4 queries, N2 keys, H=12, HEAD=64).
// One block per (batch, head). KV interleaved bf16: row (b*N2+j), k at col
// h*64+d, v at col 768+h*64+d.  O fp32: (b*4+i, h*64+d).
// ---------------------------------------------------------------------------
__global__ __launch_bounds__(256) void cross_attn_kernel(
    const float* __restrict__ Q, const __bf16* __restrict__ KV,
    int N2, float* __restrict__ O)
{
    __shared__ float sc[4][800];
    __shared__ float qs[4][64];
    __shared__ float red[4][64];

    const int blk = blockIdx.x;
    const int b = blk / 12, h = blk % 12;
    const int tid = threadIdx.x;

    { // load q (4x64)
        const int i = tid >> 6, d = tid & 63;
        qs[i][d] = Q[((size_t)(b * 4 + i)) * 768 + h * 64 + d];
    }
    __syncthreads();

    // scores = q @ k^T * SCALE
    for (int idx = tid; idx < 4 * N2; idx += 256) {
        const int i = idx / N2, j = idx % N2;
        const __bf16* kr = KV + ((size_t)b * N2 + j) * 1536 + h * 64;
        float d = 0.0f;
#pragma unroll 8
        for (int t = 0; t < 64; ++t) d += qs[i][t] * (float)kr[t];
        sc[i][j] = d * 0.125f;
    }
    __syncthreads();

    // rowwise softmax (64 threads per row)
    const int row = tid >> 6, l = tid & 63;
    float m = -1e30f;
    for (int j = l; j < N2; j += 64) m = fmaxf(m, sc[row][j]);
    red[row][l] = m;
    __syncthreads();
    for (int s = 32; s > 0; s >>= 1) {
        if (l < s) red[row][l] = fmaxf(red[row][l], red[row][l + s]);
        __syncthreads();
    }
    const float rmax = red[row][0];
    __syncthreads();
    float sum = 0.0f;
    for (int j = l; j < N2; j += 64) {
        const float e = __expf(sc[row][j] - rmax);
        sc[row][j] = e;
        sum += e;
    }
    red[row][l] = sum;
    __syncthreads();
    for (int s = 32; s > 0; s >>= 1) {
        if (l < s) red[row][l] += red[row][l + s];
        __syncthreads();
    }
    const float inv = 1.0f / red[row][0];
    __syncthreads();
    for (int j = l; j < N2; j += 64) sc[row][j] *= inv;
    __syncthreads();

    // o = attn @ v  (thread -> (query i, dim d))
    {
        const int i = tid >> 6, d0 = tid & 63;
        const __bf16* vb = KV + ((size_t)b * N2) * 1536 + 768 + h * 64 + d0;
        float acc = 0.0f;
        for (int j = 0; j < N2; ++j) acc += sc[i][j] * (float)vb[(size_t)j * 1536];
        O[((size_t)(b * 4 + i)) * 768 + h * 64 + d0] = acc;
    }
}

// ---------------------------------------------------------------------------
// Fused attention core (N1=8, N2=16) with separate fp32 K/V and attn output.
// One block per (batch, head). Writes attn (B,H,8,16) and O (b*8+i, h*64+d).
// ---------------------------------------------------------------------------
__global__ __launch_bounds__(256) void fused_attn_kernel(
    const float* __restrict__ Q, const float* __restrict__ K2,
    const float* __restrict__ V2,
    float* __restrict__ attn_out, float* __restrict__ O)
{
    __shared__ float sc[8][16];
    const int blk = blockIdx.x;
    const int b = blk / 12, h = blk % 12;
    const int tid = threadIdx.x;

    if (tid < 128) {
        const int i = tid >> 4, j = tid & 15;
        const float* q = Q + ((size_t)(b * 8 + i)) * 768 + h * 64;
        const float* k = K2 + ((size_t)(b * 16 + j)) * 768 + h * 64;
        float d = 0.0f;
#pragma unroll
        for (int t = 0; t < 64; ++t) d += q[t] * k[t];
        sc[i][j] = d * 0.125f;
    }
    __syncthreads();
    if (tid < 8) {
        float mx = sc[tid][0];
#pragma unroll
        for (int j = 1; j < 16; ++j) mx = fmaxf(mx, sc[tid][j]);
        float s = 0.0f;
#pragma unroll
        for (int j = 0; j < 16; ++j) { const float e = __expf(sc[tid][j] - mx); sc[tid][j] = e; s += e; }
        const float inv = 1.0f / s;
#pragma unroll
        for (int j = 0; j < 16; ++j) sc[tid][j] *= inv;
    }
    __syncthreads();
    if (tid < 128) {
        const int i = tid >> 4, j = tid & 15;
        attn_out[(((size_t)(b * 12 + h)) * 8 + i) * 16 + j] = sc[i][j];
    }
    for (int idx = tid; idx < 512; idx += 256) {
        const int i = idx >> 6, d0 = idx & 63;
        float acc = 0.0f;
#pragma unroll
        for (int j = 0; j < 16; ++j)
            acc += sc[i][j] * V2[((size_t)(b * 16 + j)) * 768 + h * 64 + d0];
        O[((size_t)(b * 8 + i)) * 768 + h * 64 + d0] = acc;
    }
}

// ---------------------------------------------------------------------------
// xva[b, i*4+j, :768] = xmm_v[b,i];  xva[b, i*4+j, 768:] = xmm_a[b,j]
// ---------------------------------------------------------------------------
__global__ void build_xva_kernel(const float* __restrict__ xmm_v,
                                 const float* __restrict__ xmm_a,
                                 float* __restrict__ xva, int total)
{
    for (int g = blockIdx.x * blockDim.x + threadIdx.x; g < total;
         g += gridDim.x * blockDim.x) {
        const int b = g / (16 * 1536);
        const int rem = g % (16 * 1536);
        const int mrow = rem / 1536, c = rem % 1536;
        const int i = mrow >> 2, j = mrow & 3;
        xva[g] = (c < 768) ? xmm_v[((size_t)(b * 4 + i)) * 768 + c]
                           : xmm_a[((size_t)(b * 4 + j)) * 768 + (c - 768)];
    }
}

// xmm_out = concat([o_fused(8), xmm_v(4), xmm_a(4)], token axis)
__global__ void assemble_out_kernel(const float* __restrict__ of,
                                    const float* __restrict__ xmm_v,
                                    const float* __restrict__ xmm_a,
                                    float* __restrict__ out, int total)
{
    for (int g = blockIdx.x * blockDim.x + threadIdx.x; g < total;
         g += gridDim.x * blockDim.x) {
        const int b = g / (16 * 768);
        const int rem = g % (16 * 768);
        const int r = rem / 768, c = rem % 768;
        float v;
        if (r < 8)       v = of[((size_t)(b * 8 + r)) * 768 + c];
        else if (r < 12) v = xmm_v[((size_t)(b * 4 + r - 8)) * 768 + c];
        else             v = xmm_a[((size_t)(b * 4 + r - 12)) * 768 + c];
        out[g] = v;
    }
}

// ---------------------------------------------------------------------------
// Host orchestration
// ---------------------------------------------------------------------------
extern "C" void kernel_launch(void* const* d_in, const int* in_sizes, int n_in,
                              void* d_out, int out_size, void* d_ws, size_t ws_size,
                              hipStream_t stream)
{
    const float* xmm     = (const float*)d_in[0];
    const float* xv      = (const float*)d_in[1];
    const float* xa      = (const float*)d_in[2];
    const float* qv_w    = (const float*)d_in[3];
    const float* kvv_w   = (const float*)d_in[4];
    const float* projv_w = (const float*)d_in[5];
    const float* projv_b = (const float*)d_in[6];
    const float* qa_w    = (const float*)d_in[7];
    const float* kva_w   = (const float*)d_in[8];
    const float* proja_w = (const float*)d_in[9];
    const float* proja_b = (const float*)d_in[10];
    const float* fq_w    = (const float*)d_in[11];
    const float* fk_w    = (const float*)d_in[12];
    const float* fv_w    = (const float*)d_in[13];
    const float* fproj_w = (const float*)d_in[14];
    const float* fproj_b = (const float*)d_in[15];
    float* out = (float*)d_out;

    char* ws = (char*)d_ws;
    size_t off = 0;
    auto alloc = [&](size_t bytes) -> void* {
        void* p = ws + off;
        off = (off + bytes + 255) & ~(size_t)255;
        return p;
    };

    __bf16* KVv   = (__bf16*)alloc((size_t)64 * 784 * 1536 * 2); // bf16 KV, V branch
    __bf16* KVa   = (__bf16*)alloc((size_t)64 * 512 * 1536 * 2); // bf16 KV, A branch
    float* qv_f   = (float*)alloc((size_t)256 * 768 * 4);
    float* qa_f   = (float*)alloc((size_t)256 * 768 * 4);
    float* Ov_f   = (float*)alloc((size_t)256 * 768 * 4);
    float* Oa_f   = (float*)alloc((size_t)256 * 768 * 4);
    float* xmmv_f = (float*)alloc((size_t)256 * 768 * 4);
    float* xmma_f = (float*)alloc((size_t)256 * 768 * 4);
    float* xva_f  = (float*)alloc((size_t)1024 * 1536 * 4);
    float* q2_f   = (float*)alloc((size_t)512 * 768 * 4);
    float* k2_f   = (float*)alloc((size_t)1024 * 768 * 4);
    float* v2_f   = (float*)alloc((size_t)1024 * 768 * 4);
    float* o2_f   = (float*)alloc((size_t)512 * 768 * 4);
    float* of_f   = (float*)alloc((size_t)512 * 768 * 4);
    // pre-transposed bf16 weights (WT: N x K row-major)
    __bf16* qvT   = (__bf16*)alloc((size_t)768 * 768 * 2);
    __bf16* qaT   = (__bf16*)alloc((size_t)768 * 768 * 2);
    __bf16* kvvT  = (__bf16*)alloc((size_t)1536 * 768 * 2);
    __bf16* kvaT  = (__bf16*)alloc((size_t)1536 * 768 * 2);
    __bf16* projvT= (__bf16*)alloc((size_t)768 * 768 * 2);
    __bf16* projaT= (__bf16*)alloc((size_t)768 * 768 * 2);
    __bf16* fqT   = (__bf16*)alloc((size_t)768 * 768 * 2);
    __bf16* fkT   = (__bf16*)alloc((size_t)768 * 1536 * 2);
    __bf16* fvT   = (__bf16*)alloc((size_t)768 * 1536 * 2);
    __bf16* fprojT= (__bf16*)alloc((size_t)768 * 768 * 2);
    (void)ws_size; (void)in_sizes; (void)n_in; (void)out_size;

    const dim3 blk(256);
    const int IDG = 1 << 30;   // identity row-map

    // 0) Weight pre-pass: transpose + convert to bf16 (few MB, one-time)
    auto tp = [&](const float* W, __bf16* WT, int K, int N) {
        const int total = K * N;
        transpose_to_bf16_kernel<<<dim3((total + 255) / 256), blk, 0, stream>>>(
            W, WT, K, N, total);
    };
    tp(qv_w,    qvT,    768, 768);
    tp(qa_w,    qaT,    768, 768);
    tp(kvv_w,   kvvT,   768, 1536);
    tp(kva_w,   kvaT,   768, 1536);
    tp(projv_w, projvT, 768, 768);
    tp(proja_w, projaT, 768, 768);
    tp(fq_w,    fqT,    768, 768);
    tp(fk_w,    fkT,    1536, 768);
    tp(fv_w,    fvT,    1536, 768);
    tp(fproj_w, fprojT, 768, 768);

    // 1) Dominant KV projection GEMMs -> bf16 KV buffers
    gemm_wmma_bf16<<<dim3(392, 12), blk, 0, stream>>>(xv, kvvT, 50176, 1536, 768,
                                                      IDG, 0, 0, nullptr, nullptr, KVv);
    gemm_wmma_bf16<<<dim3(256, 12), blk, 0, stream>>>(xa, kvaT, 32768, 1536, 768,
                                                      IDG, 0, 0, nullptr, nullptr, KVa);
    // 2) Query projections (strided slices of xmm, rows 8..11 / 12..15)
    gemm_wmma_bf16<<<dim3(2, 6), blk, 0, stream>>>(xmm, qvT, 256, 768, 768,
                                                   4, 16, 8, nullptr, qv_f, nullptr);
    gemm_wmma_bf16<<<dim3(2, 6), blk, 0, stream>>>(xmm, qaT, 256, 768, 768,
                                                   4, 16, 12, nullptr, qa_f, nullptr);
    // 3) Cross attention cores (per batch,head)
    cross_attn_kernel<<<64 * 12, blk, 0, stream>>>(qv_f, KVv, 784, Ov_f);
    cross_attn_kernel<<<64 * 12, blk, 0, stream>>>(qa_f, KVa, 512, Oa_f);
    // 4) Output projections (+bias)
    gemm_wmma_bf16<<<dim3(2, 6), blk, 0, stream>>>(Ov_f, projvT, 256, 768, 768,
                                                   IDG, 0, 0, projv_b, xmmv_f, nullptr);
    gemm_wmma_bf16<<<dim3(2, 6), blk, 0, stream>>>(Oa_f, projaT, 256, 768, 768,
                                                   IDG, 0, 0, proja_b, xmma_f, nullptr);
    // 5) Fused branch
    build_xva_kernel<<<dim3((1024 * 1536 + 255) / 256), blk, 0, stream>>>(
        xmmv_f, xmma_f, xva_f, 1024 * 1536);
    gemm_wmma_bf16<<<dim3(4, 6), blk, 0, stream>>>(xmm, fqT, 512, 768, 768,
                                                   8, 16, 0, nullptr, q2_f, nullptr);
    gemm_wmma_bf16<<<dim3(8, 6), blk, 0, stream>>>(xva_f, fkT, 1024, 768, 1536,
                                                   IDG, 0, 0, nullptr, k2_f, nullptr);
    gemm_wmma_bf16<<<dim3(8, 6), blk, 0, stream>>>(xva_f, fvT, 1024, 768, 1536,
                                                   IDG, 0, 0, nullptr, v2_f, nullptr);
    fused_attn_kernel<<<64 * 12, blk, 0, stream>>>(q2_f, k2_f, v2_f,
                                                   out + 786432, o2_f);
    gemm_wmma_bf16<<<dim3(4, 6), blk, 0, stream>>>(o2_f, fprojT, 512, 768, 768,
                                                   IDG, 0, 0, fproj_b, of_f, nullptr);
    // 6) Assemble xmm_out
    assemble_out_kernel<<<dim3((786432 + 255) / 256), blk, 0, stream>>>(
        of_f, xmmv_f, xmma_f, out, 786432);
}
